// IncrementalLoRALinear_87222195847797
// MI455X (gfx1250) — matrix operand order repaired and checked by reading, not compiled, not run on hardware
//
#include <hip/hip_runtime.h>
#include <hip/hip_bf16.h>

typedef __attribute__((ext_vector_type(16))) __bf16 v16bf;
typedef __attribute__((ext_vector_type(8)))  __bf16 v8bf;
typedef __attribute__((ext_vector_type(4)))  __bf16 v4bf;
typedef __attribute__((ext_vector_type(8)))  float  v8f;

static constexpr int Mdim = 16384;   // B*S = 8*2048
static constexpr int Ndim = 4096;    // D_OUT
static constexpr int Kdim = 4096;    // D_IN
static constexpr int Tn   = 8;       // tasks
static constexpr int Rr   = 32;      // LoRA rank
static constexpr float SCALING = 1.0f;  // R / ALPHA

static constexpr int BM = 128, BN = 128, BK = 32;
static constexpr int NK = Kdim / BK;
static constexpr int LSTR = 40;      // 32 + 8 bf16 pad -> 80B row stride, bank-conflict-free

union V16U { v16bf v; v8bf h[2]; };

// ---- gfx1250 async global->LDS path (ASYNCcnt), with safe fallback --------
#if defined(__HIP_DEVICE_COMPILE__) && defined(__has_builtin)
#  if __has_builtin(__builtin_amdgcn_global_load_async_to_lds_b128)
#    define HAVE_ASYNC_LDS 1
#  endif
#  if __has_builtin(__builtin_amdgcn_s_wait_asynccnt)
#    define HAVE_WAIT_ASYNC_BUILTIN 1
#  endif
#endif
#ifndef HAVE_ASYNC_LDS
#  define HAVE_ASYNC_LDS 0
#endif
#ifndef HAVE_WAIT_ASYNC_BUILTIN
#  define HAVE_WAIT_ASYNC_BUILTIN 0
#endif

// exact element type from the clang diagnostic: int __vector_size__(16)
typedef int v4i_vs __attribute__((__vector_size__(4 * sizeof(int))));
typedef __attribute__((address_space(1))) v4i_vs* as1_v4i;
typedef __attribute__((address_space(3))) v4i_vs* as3_v4i;

__device__ __forceinline__ void copy16_g2lds(const void* g, void* l) {
#if HAVE_ASYNC_LDS
  __builtin_amdgcn_global_load_async_to_lds_b128(
      (as1_v4i)(unsigned long long)(uintptr_t)g,
      (as3_v4i)(unsigned int)(uintptr_t)l, 0, 0);
#else
  *(v8bf*)l = *(const v8bf*)g;  // via VGPRs (loadcnt + dscnt)
#endif
}

__device__ __forceinline__ void wait_async_lds() {
#if HAVE_ASYNC_LDS
#  if HAVE_WAIT_ASYNC_BUILTIN
  __builtin_amdgcn_s_wait_asynccnt(0);
#  else
  asm volatile("s_wait_asynccnt 0x0" ::: "memory");
#  endif
#endif
}

__device__ __forceinline__ void split4(const float4 v, v4bf& hv, v4bf& lv) {
  hv[0] = (__bf16)v.x; lv[0] = (__bf16)(v.x - (float)hv[0]);
  hv[1] = (__bf16)v.y; lv[1] = (__bf16)(v.y - (float)hv[1]);
  hv[2] = (__bf16)v.z; lv[2] = (__bf16)(v.z - (float)hv[2]);
  hv[3] = (__bf16)v.w; lv[3] = (__bf16)(v.w - (float)hv[3]);
}

// ---------------------------------------------------------------------------
// Prep 1: W_eff[o][d] = W[o][d] + SCALING * sum_t sum_r A[t][d][r]*B[t][o][r]
// stored as split bf16 (hi, lo).  o = blockIdx.y (B rows become uniform /
// scalar loads), d = blockIdx.x*256 + tid (A loads and stores coalesced).
// ---------------------------------------------------------------------------
__global__ __launch_bounds__(256) void lora_fold_kernel(
    const float* __restrict__ W, const float* __restrict__ As,
    const float* __restrict__ Bs, unsigned short* __restrict__ whi_u,
    unsigned short* __restrict__ wlo_u) {
  __bf16* whi = reinterpret_cast<__bf16*>(whi_u);
  __bf16* wlo = reinterpret_cast<__bf16*>(wlo_u);
  const int d = blockIdx.x * 256 + threadIdx.x;
  const int o = blockIdx.y;

  float s = 0.f;
#pragma unroll
  for (int t = 0; t < Tn; ++t) {
    const float* a = As + ((size_t)t * Kdim + d) * Rr;   // A[t][d][*]
    const float* b = Bs + ((size_t)t * Ndim + o) * Rr;   // B[t][o][*] (uniform)
#pragma unroll
    for (int r = 0; r < Rr; r += 4) {
      const float4 av = *(const float4*)(a + r);
      const float4 bv = *(const float4*)(b + r);
      s += av.x * bv.x + av.y * bv.y + av.z * bv.z + av.w * bv.w;
    }
  }
  const size_t idx = (size_t)o * Kdim + d;
  const float e = W[idx] + SCALING * s;
  const __bf16 h = (__bf16)e;
  whi[idx] = h;
  wlo[idx] = (__bf16)(e - (float)h);
}

// ---------------------------------------------------------------------------
// Prep 2: pre-split x (f32) into bf16 hi/lo arrays once, so the GEMM's A
// tiles become pure async byte-copies (no in-loop conversion VALU, no
// WMMA->VALU hazards).  Streams 256 MB in + 256 MB out (~22 us at 23.3 TB/s).
// ---------------------------------------------------------------------------
__global__ __launch_bounds__(256) void x_split_kernel(
    const float* __restrict__ x, unsigned short* __restrict__ xhi_u,
    unsigned short* __restrict__ xlo_u) {
  __bf16* xh = reinterpret_cast<__bf16*>(xhi_u);
  __bf16* xl = reinterpret_cast<__bf16*>(xlo_u);
  const size_t i = ((size_t)blockIdx.x * 256 + threadIdx.x) * 4;
  const float4 v = *(const float4*)(x + i);
  v4bf hv, lv;
  split4(v, hv, lv);
  *(v4bf*)(xh + i) = hv;
  *(v4bf*)(xl + i) = lv;
}

// ---------------------------------------------------------------------------
// Main GEMM: out = x @ W_eff^T + bias, split-bf16 WMMA (xh*wh + xh*wl + xl*wh)
// 256 threads = 8 waves; block tile 128x128; each wave 32x64 (2x4 C-tiles).
// Double-buffered LDS (80 KB), async global->LDS staging, one barrier per
// K-step.  PRESPLIT: A tiles async-copied from pre-split xh/xl; otherwise
// x is converted in-flight through VGPRs.
// ---------------------------------------------------------------------------
template <bool PRESPLIT>
__global__ __launch_bounds__(256) void lora_wmma_gemm_kernel(
    const float* __restrict__ x,
    const unsigned short* __restrict__ xhi_u, const unsigned short* __restrict__ xlo_u,
    const unsigned short* __restrict__ whi_u, const unsigned short* __restrict__ wlo_u,
    const float* __restrict__ bias, float* __restrict__ out) {
  const __bf16* xhi = reinterpret_cast<const __bf16*>(xhi_u);
  const __bf16* xlo = reinterpret_cast<const __bf16*>(xlo_u);
  const __bf16* whi = reinterpret_cast<const __bf16*>(whi_u);
  const __bf16* wlo = reinterpret_cast<const __bf16*>(wlo_u);

  __shared__ __align__(16) __bf16 Ah[2][BM * LSTR];
  __shared__ __align__(16) __bf16 Al[2][BM * LSTR];
  __shared__ __align__(16) __bf16 Bh[2][BN * LSTR];
  __shared__ __align__(16) __bf16 Bl[2][BN * LSTR];

  const int tid  = threadIdx.x;
  const int lane = tid & 31;
  const int wave = tid >> 5;
  const int wm = (wave & 3) * 32;   // wave M offset in block tile
  const int wn = (wave >> 2) * 64;  // wave N offset in block tile
  const int bm = blockIdx.y * BM;
  const int bn = blockIdx.x * BN;

  // ISA 7.12.2 fragment addressing (wave32, 16-bit A 16x32 / B 32x16)
  const int ar = lane & 15;           // A: row M
  const int ak = (lane >> 4) << 3;    // A: K sub-offset 0 or 8
  const int bc = lane & 15;           // B: column N
  const int bk = (lane >> 4) << 3;    // B: K sub-offset 0 or 8

  v8f acc[2][4] = {};
  float4 xv[4];   // x prefetch registers (non-presplit path only)

  // issue global traffic for stage (buf, k0)
  auto issue_stage = [&](int buf, int k0) {
#pragma unroll
    for (int it = 0; it < 2; ++it) {
      const int i   = tid + it * 256;          // 0..511 chunks of 8 bf16
      const int row = i >> 2;
      const int c8  = (i & 3) << 3;
      const size_t g = (size_t)(bn + row) * Kdim + k0 + c8;
      copy16_g2lds(whi + g, &Bh[buf][row * LSTR + c8]);
      copy16_g2lds(wlo + g, &Bl[buf][row * LSTR + c8]);
    }
    if constexpr (PRESPLIT) {
#pragma unroll
      for (int it = 0; it < 2; ++it) {
        const int i   = tid + it * 256;
        const int row = i >> 2;
        const int c8  = (i & 3) << 3;
        const size_t g = (size_t)(bm + row) * Kdim + k0 + c8;
        copy16_g2lds(xhi + g, &Ah[buf][row * LSTR + c8]);
        copy16_g2lds(xlo + g, &Al[buf][row * LSTR + c8]);
      }
    } else {
#pragma unroll
      for (int it = 0; it < 4; ++it) {
        const int i   = tid + it * 256;        // 0..1023 float4 chunks
        const int row = i >> 3;
        const int c4  = (i & 7) << 2;
        xv[it] = *(const float4*)(x + (size_t)(bm + row) * Kdim + k0 + c4);
      }
    }
  };
  // convert prefetched x to split bf16 and store into LDS (non-presplit)
  auto store_stage_x = [&](int buf) {
#pragma unroll
    for (int it = 0; it < 4; ++it) {
      const int i   = tid + it * 256;
      const int row = i >> 3;
      const int c4  = (i & 7) << 2;
      v4bf hv, lv;
      split4(xv[it], hv, lv);
      *(v4bf*)&Ah[buf][row * LSTR + c4] = hv;
      *(v4bf*)&Al[buf][row * LSTR + c4] = lv;
    }
  };

  // ---- prologue: fill stage 0 ----
  issue_stage(0, 0);
  if constexpr (!PRESPLIT) store_stage_x(0);
  wait_async_lds();
  __syncthreads();

  for (int kk = 0; kk < NK; ++kk) {
    const int cur  = kk & 1;
    const bool more = (kk + 1) < NK;
    if (more) issue_stage(cur ^ 1, (kk + 1) * BK);   // overlap with compute

    // ---- load fragments per ISA VGPR layout (stage cur) ----
    V16U afh[2], afl[2];
#pragma unroll
    for (int mi = 0; mi < 2; ++mi) {
      const __bf16* p = &Ah[cur][(wm + mi * 16 + ar) * LSTR];
      afh[mi].h[0] = *(const v8bf*)(p + ak);
      afh[mi].h[1] = *(const v8bf*)(p + 16 + ak);
      const __bf16* q = &Al[cur][(wm + mi * 16 + ar) * LSTR];
      afl[mi].h[0] = *(const v8bf*)(q + ak);
      afl[mi].h[1] = *(const v8bf*)(q + 16 + ak);
    }
    V16U bfh[4], bfl[4];
#pragma unroll
    for (int ni = 0; ni < 4; ++ni) {
      const __bf16* p = &Bh[cur][(wn + ni * 16 + bc) * LSTR + bk];
      bfh[ni].h[0] = *(const v8bf*)(p);
      bfh[ni].h[1] = *(const v8bf*)(p + 16);
      const __bf16* q = &Bl[cur][(wn + ni * 16 + bc) * LSTR + bk];
      bfl[ni].h[0] = *(const v8bf*)(q);
      bfl[ni].h[1] = *(const v8bf*)(q + 16);
    }

    // ---- 3-product split-bf16 accumulation (fp32-accurate) ----
#pragma unroll
    for (int mi = 0; mi < 2; ++mi) {
#pragma unroll
      for (int ni = 0; ni < 4; ++ni) {
        acc[mi][ni] = __builtin_amdgcn_wmma_f32_16x16x32_bf16(
            false, afh[mi].v, false, bfh[ni].v, (short)0, acc[mi][ni], false, false);
        acc[mi][ni] = __builtin_amdgcn_wmma_f32_16x16x32_bf16(
            false, afh[mi].v, false, bfl[ni].v, (short)0, acc[mi][ni], false, false);
        acc[mi][ni] = __builtin_amdgcn_wmma_f32_16x16x32_bf16(
            false, afl[mi].v, false, bfh[ni].v, (short)0, acc[mi][ni], false, false);
      }
    }

    if constexpr (!PRESPLIT) {
      if (more) store_stage_x(cur ^ 1);  // convert+store x behind the WMMAs
    }
    wait_async_lds();
    __syncthreads();                     // single barrier per K-step
  }

  // ---- epilogue: C layout (VGPR j = M row; lane%16 = N; lane>=16 -> M+8) ----
  const int cn = lane & 15;
  const int cm = (lane >> 4) << 3;
#pragma unroll
  for (int mi = 0; mi < 2; ++mi) {
#pragma unroll
    for (int ni = 0; ni < 4; ++ni) {
      const int gn = bn + wn + ni * 16 + cn;
      const float bv = bias[gn];
      const size_t base = (size_t)(bm + wm + mi * 16 + cm) * Ndim + gn;
#pragma unroll
      for (int j = 0; j < 8; ++j)
        out[base + (size_t)j * Ndim] = acc[mi][ni][j] + bv;
    }
  }
}

// ---------------------------------------------------------------------------
extern "C" void kernel_launch(void* const* d_in, const int* in_sizes, int n_in,
                              void* d_out, int out_size, void* d_ws, size_t ws_size,
                              hipStream_t stream) {
  const float* x    = (const float*)d_in[0];
  const float* w    = (const float*)d_in[1];
  const float* bias = (const float*)d_in[2];
  const float* As   = (const float*)d_in[3];
  const float* Bs   = (const float*)d_in[4];
  float* out = (float*)d_out;

  // scratch layout: [whi 32MB][wlo 32MB][xhi 128MB][xlo 128MB... ] (bf16)
  const size_t wElems = (size_t)Ndim * Kdim;       // 16.7M
  const size_t xElems = (size_t)Mdim * Kdim;       // 67.1M
  const size_t needW  = 2 * wElems * sizeof(unsigned short);            // 64 MB
  const size_t needWX = needW + 2 * xElems * sizeof(unsigned short);    // 320 MB

  unsigned short* whi = (unsigned short*)d_ws;
  unsigned short* wlo = whi + wElems;
  unsigned short* xhi = wlo + wElems;
  unsigned short* xlo = xhi + xElems;

  lora_fold_kernel<<<dim3(Kdim / 256, Ndim), 256, 0, stream>>>(w, As, Bs, whi, wlo);

  if (ws_size >= needWX) {
    // fast path: pre-split x once; GEMM stages A and B tiles purely via
    // async global->LDS (no in-loop conversions)
    x_split_kernel<<<dim3((unsigned)(xElems / 4 / 256)), 256, 0, stream>>>(x, xhi, xlo);
    lora_wmma_gemm_kernel<true><<<dim3(Ndim / BN, Mdim / BM), 256, 0, stream>>>(
        x, xhi, xlo, whi, wlo, bias, out);
  } else {
    // fallback: convert x in-flight (needs only 64 MB scratch)
    lora_wmma_gemm_kernel<false><<<dim3(Ndim / BN, Mdim / BM), 256, 0, stream>>>(
        x, xhi, xlo, whi, wlo, bias, out);
  }
}
